// EncoderAndMemory_4054449127999
// MI455X (gfx1250) — matrix operand order repaired and checked by reading, not compiled, not run on hardware
//
#include <hip/hip_runtime.h>

// ---------------------------------------------------------------------------
// EncoderAndMemory (MemAE-style): cosine-sim logits -> softmax -> hard shrink
// -> L1 normalize -> reconstruction.
// GEMMs: V_WMMA_F32_16X16X32_BF16 with LDS tiles staged via
// GLOBAL_LOAD_ASYNC_TO_LDS_B128 (ASYNCcnt double-buffered pipeline).
// ---------------------------------------------------------------------------

typedef __attribute__((ext_vector_type(16))) __bf16 v16bf;
typedef __attribute__((ext_vector_type(8)))  float  v8f;

#define BATCH   8192
#define DDIM    4096
#define MDIM    600
#define MPAD    640          // 10 tiles of 64; also K-pad for GEMM2 (20*32)
#define THRESH  (1.0f / 600.0f)
#define LDSPAD  40           // staged-tile row pitch in bf16 (80B, bank-skewed)

__device__ __forceinline__ unsigned short f2bf(float f) {
    unsigned int u = __float_as_uint(f);
    unsigned int r = u + 0x7FFFu + ((u >> 16) & 1u);   // round-to-nearest-even
    return (unsigned short)(r >> 16);
}
__device__ __forceinline__ unsigned int pack2bf(float a, float b) {
    return (unsigned int)f2bf(a) | ((unsigned int)f2bf(b) << 16);
}

// A-fragment 16x32 bf16: lane r=lane&15 holds row M=r;
// half=lane>>4 selects K {0..15} vs {16..31} (contiguous, matches B).
// p points at row_base + half*16 elements; second b128 at +8 elements.
__device__ __forceinline__ v16bf load_frag_a(const unsigned short* p) {
    union { v16bf v; uint4 u[2]; } t;
    t.u[0] = *reinterpret_cast<const uint4*>(p);
    t.u[1] = *reinterpret_cast<const uint4*>(p + 8);
    return t.v;
}
// B-fragment 32x16 bf16: lane holds column N=lane&15; 16 contiguous K values
// starting at half*16.  p points at col_row + half*16 elements.
__device__ __forceinline__ v16bf load_frag_b(const unsigned short* p) {
    union { v16bf v; uint4 u[2]; } t;
    t.u[0] = *reinterpret_cast<const uint4*>(p);
    t.u[1] = *reinterpret_cast<const uint4*>(p + 8);
    return t.v;
}

__device__ __forceinline__ v8f wmma_bf16(v16bf a, v16bf b, v8f c) {
    return __builtin_amdgcn_wmma_f32_16x16x32_bf16(
        false, a, false, b, (short)0, c, false, false);
}

// CDNA5 async copy: 16 bytes global -> LDS per lane, tracked by ASYNCcnt.
__device__ __forceinline__ void async_b128(unsigned lds_off, const void* gptr) {
    asm volatile("global_load_async_to_lds_b128 %0, %1, off"
                 :: "v"(lds_off), "v"(gptr) : "memory");
}
// Generic LDS pointers carry the LDS byte offset in their low 32 bits.
__device__ __forceinline__ unsigned lds_off_of(const void* p) {
    return (unsigned)(unsigned long long)p;
}

// ---------------------------------------------------------------------------
// Kernel 1: per-row L2 norm of z + bf16 copy.  One block per row (4096 elems).
// ---------------------------------------------------------------------------
__global__ __launch_bounds__(256) void prep_z_kernel(
        const float* __restrict__ x, unsigned short* __restrict__ zb,
        float* __restrict__ znorm) {
    const int row = blockIdx.x;
    const int t   = threadIdx.x;
    const float* src = x + (size_t)row * DDIM;
    unsigned short* dst = zb + (size_t)row * DDIM;
    float s = 0.f;
#pragma unroll
    for (int i = 0; i < 4; ++i) {
        float4 v = reinterpret_cast<const float4*>(src)[t + 256 * i];
        s += v.x * v.x + v.y * v.y + v.z * v.z + v.w * v.w;
        uint2 p;
        p.x = pack2bf(v.x, v.y);
        p.y = pack2bf(v.z, v.w);
        reinterpret_cast<uint2*>(dst)[t + 256 * i] = p;
    }
    __shared__ float red[256];
    red[t] = s;
    __syncthreads();
    for (int off = 128; off > 0; off >>= 1) {
        if (t < off) red[t] += red[t + off];
        __syncthreads();
    }
    if (t == 0) znorm[row] = sqrtf(red[0]);
}

// ---------------------------------------------------------------------------
// Kernel 2: memory rows -> bf16 (row-major, MPAD rows) + bf16 transpose
// (DDIM x MPAD) + per-row norms.  Rows >= MDIM are zero padding.
// ---------------------------------------------------------------------------
__global__ __launch_bounds__(256) void prep_mem_kernel(
        const float* __restrict__ mem, unsigned short* __restrict__ mb,
        unsigned short* __restrict__ mTb, float* __restrict__ mnorm) {
    const int row = blockIdx.x;     // 0..MPAD-1
    const int t   = threadIdx.x;
    unsigned short* dst = mb + (size_t)row * DDIM;
    if (row < MDIM) {
        const float* src = mem + (size_t)row * DDIM;
        float s = 0.f;
#pragma unroll
        for (int i = 0; i < 4; ++i) {
            const int e4 = t + 256 * i;             // float4 index
            float4 v = reinterpret_cast<const float4*>(src)[e4];
            s += v.x * v.x + v.y * v.y + v.z * v.z + v.w * v.w;
            uint2 p;
            p.x = pack2bf(v.x, v.y);
            p.y = pack2bf(v.z, v.w);
            reinterpret_cast<uint2*>(dst)[e4] = p;
            const int e = e4 * 4;                    // scalar index
            mTb[(size_t)(e + 0) * MPAD + row] = f2bf(v.x);
            mTb[(size_t)(e + 1) * MPAD + row] = f2bf(v.y);
            mTb[(size_t)(e + 2) * MPAD + row] = f2bf(v.z);
            mTb[(size_t)(e + 3) * MPAD + row] = f2bf(v.w);
        }
        __shared__ float red[256];
        red[t] = s;
        __syncthreads();
        for (int off = 128; off > 0; off >>= 1) {
            if (t < off) red[t] += red[t + off];
            __syncthreads();
        }
        if (t == 0) mnorm[row] = sqrtf(red[0]);
    } else {
#pragma unroll
        for (int i = 0; i < 4; ++i) {
            const int e4 = t + 256 * i;
            reinterpret_cast<uint2*>(dst)[e4] = make_uint2(0u, 0u);
            const int e = e4 * 4;
            mTb[(size_t)(e + 0) * MPAD + row] = 0;
            mTb[(size_t)(e + 1) * MPAD + row] = 0;
            mTb[(size_t)(e + 2) * MPAD + row] = 0;
            mTb[(size_t)(e + 3) * MPAD + row] = 0;
        }
        if (t == 0) mnorm[row] = 1.0f;
    }
}

// ---------------------------------------------------------------------------
// Shared GEMM core layout: block tile 128(M) x 64(N), 8 waves (4x2),
// wave tile 32x32, K step 32.  Tiles staged in LDS by async copies:
// 768 b128 chunks per stage = 3 per thread (A: 512, B: 256).
// ---------------------------------------------------------------------------

// ---------------------------------------------------------------------------
// Kernel 3: GEMM1  logits = (z @ memT) / max(|z||m|, 1e-8)
// ---------------------------------------------------------------------------
__global__ __launch_bounds__(256) void gemm1_kernel(
        const unsigned short* __restrict__ zb,
        const unsigned short* __restrict__ mb,
        const float* __restrict__ znorm, const float* __restrict__ mnorm,
        float* __restrict__ logits /* BATCH x MDIM */) {
    __shared__ unsigned short Ab[2][128][LDSPAD];
    __shared__ unsigned short Bb[2][64][LDSPAD];

    const int t    = threadIdx.x;
    const int lane = t & 31;
    const int wave = t >> 5;
    const int wm = wave >> 1;          // 0..3
    const int wn = wave & 1;           // 0..1
    const int r    = lane & 15;
    const int half = lane >> 4;
    const int rowBase0 = blockIdx.x * 128;
    const int colBase0 = blockIdx.y * 64;

    // per-thread async-copy assignment: chunks t, t+256, t+512
    const int ar0 = t >> 2,         as0 = t & 3;          // A chunk 0
    const int ar1 = (t + 256) >> 2, as1 = t & 3;          // A chunk 1
    const int br  = t >> 2,         bs  = t & 3;          // B chunk
    const unsigned short* gA0 = zb + (size_t)(rowBase0 + ar0) * DDIM + as0 * 8;
    const unsigned short* gA1 = zb + (size_t)(rowBase0 + ar1) * DDIM + as1 * 8;
    const unsigned short* gB  = mb + (size_t)(colBase0 + br ) * DDIM + bs  * 8;
    unsigned lA0[2], lA1[2], lB[2];
#pragma unroll
    for (int s = 0; s < 2; ++s) {
        lA0[s] = lds_off_of(&Ab[s][ar0][as0 * 8]);
        lA1[s] = lds_off_of(&Ab[s][ar1][as1 * 8]);
        lB[s]  = lds_off_of(&Bb[s][br ][bs  * 8]);
    }

    v8f acc[2][2];
#pragma unroll
    for (int i = 0; i < 2; ++i)
#pragma unroll
        for (int j = 0; j < 2; ++j) acc[i][j] = (v8f)(0.0f);

    // prologue: stage 0
    async_b128(lA0[0], gA0);
    async_b128(lA1[0], gA1);
    async_b128(lB[0],  gB);

    for (int k = 0; k < DDIM; k += 32) {
        const int cur = (k >> 5) & 1;
        const int nxt = cur ^ 1;
        if (k + 32 < DDIM) {
            async_b128(lA0[nxt], gA0 + k + 32);
            async_b128(lA1[nxt], gA1 + k + 32);
            async_b128(lB[nxt],  gB  + k + 32);
            asm volatile("s_wait_asynccnt 0x3" ::: "memory");  // stage `cur` done
        } else {
            asm volatile("s_wait_asynccnt 0x0" ::: "memory");
        }
        __syncthreads();   // publish stage `cur` across waves

        v16bf fa0 = load_frag_a(&Ab[cur][wm * 32 + r][half * 16]);
        v16bf fa1 = load_frag_a(&Ab[cur][wm * 32 + 16 + r][half * 16]);
        v16bf fb0 = load_frag_b(&Bb[cur][wn * 32 + r][half * 16]);
        v16bf fb1 = load_frag_b(&Bb[cur][wn * 32 + 16 + r][half * 16]);
        acc[0][0] = wmma_bf16(fa0, fb0, acc[0][0]);
        acc[0][1] = wmma_bf16(fa0, fb1, acc[0][1]);
        acc[1][0] = wmma_bf16(fa1, fb0, acc[1][0]);
        acc[1][1] = wmma_bf16(fa1, fb1, acc[1][1]);

        __syncthreads();   // all reads of `cur` done before it is overwritten
    }

#pragma unroll
    for (int j = 0; j < 2; ++j) {
        const int gj = colBase0 + wn * 32 + j * 16 + r;
        const float mn = mnorm[gj];
#pragma unroll
        for (int i = 0; i < 2; ++i) {
#pragma unroll
            for (int v = 0; v < 8; ++v) {
                const int gi = rowBase0 + wm * 32 + i * 16 + v + half * 8;
                const float zn = znorm[gi];
                const float val = acc[i][j][v] / fmaxf(zn * mn, 1e-8f);
                if (gj < MDIM)
                    logits[(size_t)gi * MDIM + gj] = val;
            }
        }
    }
}

// ---------------------------------------------------------------------------
// Kernel 4: per-row softmax -> hard shrink -> L1 normalize.
// In-place on the f32 w buffer; also emits bf16 copy padded to MPAD cols.
// ---------------------------------------------------------------------------
__global__ __launch_bounds__(256) void softmax_shrink_kernel(
        float* __restrict__ w /* BATCH x MDIM, in-place */,
        unsigned short* __restrict__ wb /* BATCH x MPAD bf16 */) {
    const int row = blockIdx.x;
    const int t   = threadIdx.x;
    float* W = w + (size_t)row * MDIM;
    __shared__ float red[256];

    float v[3];
    float mx = -3.0e38f;
#pragma unroll
    for (int c = 0; c < 3; ++c) {
        const int idx = t + c * 256;
        v[c] = (idx < MDIM) ? W[idx] : -3.0e38f;
        mx = fmaxf(mx, v[c]);
    }
    red[t] = mx; __syncthreads();
    for (int off = 128; off > 0; off >>= 1) {
        if (t < off) red[t] = fmaxf(red[t], red[t + off]);
        __syncthreads();
    }
    const float rowMax = red[0];
    __syncthreads();

    float s = 0.f;
#pragma unroll
    for (int c = 0; c < 3; ++c) {
        const int idx = t + c * 256;
        v[c] = (idx < MDIM) ? __expf(v[c] - rowMax) : 0.f;
        s += v[c];
    }
    red[t] = s; __syncthreads();
    for (int off = 128; off > 0; off >>= 1) {
        if (t < off) red[t] += red[t + off];
        __syncthreads();
    }
    const float inv = 1.0f / red[0];
    __syncthreads();

    float s2 = 0.f;
#pragma unroll
    for (int c = 0; c < 3; ++c) {
        const float wv = v[c] * inv;
        const float d  = wv - THRESH;
        const float sh = fmaxf(d, 0.f) * wv / (fabsf(d) + 1e-12f);
        v[c] = sh;
        s2 += fabsf(sh);
    }
    red[t] = s2; __syncthreads();
    for (int off = 128; off > 0; off >>= 1) {
        if (t < off) red[t] += red[t + off];
        __syncthreads();
    }
    const float inv2 = 1.0f / fmaxf(red[0], 1e-12f);

    unsigned short* WB = wb + (size_t)row * MPAD;
#pragma unroll
    for (int c = 0; c < 3; ++c) {
        const int idx = t + c * 256;
        if (idx < MDIM) {
            const float out = v[c] * inv2;
            W[idx]  = out;
            WB[idx] = f2bf(out);
        } else if (idx < MPAD) {
            WB[idx] = 0;                // zero K-padding for GEMM2
        }
    }
}

// ---------------------------------------------------------------------------
// Kernel 5: GEMM2  z_hat = w(BATCH x MPAD) @ memory(MPAD x DDIM)
// A = wb (ld MPAD), B via transposed bf16 copy mTb (ld MPAD).
// ---------------------------------------------------------------------------
__global__ __launch_bounds__(256) void gemm2_kernel(
        const unsigned short* __restrict__ wb,
        const unsigned short* __restrict__ mTb,
        float* __restrict__ zhat /* BATCH x DDIM */) {
    __shared__ unsigned short Ab[2][128][LDSPAD];
    __shared__ unsigned short Bb[2][64][LDSPAD];

    const int t    = threadIdx.x;
    const int lane = t & 31;
    const int wave = t >> 5;
    const int wm = wave >> 1;
    const int wn = wave & 1;
    const int r    = lane & 15;
    const int half = lane >> 4;
    const int rowBase0 = blockIdx.x * 128;
    const int colBase0 = blockIdx.y * 64;

    const int ar0 = t >> 2,         as0 = t & 3;
    const int ar1 = (t + 256) >> 2, as1 = t & 3;
    const int br  = t >> 2,         bs  = t & 3;
    const unsigned short* gA0 = wb  + (size_t)(rowBase0 + ar0) * MPAD + as0 * 8;
    const unsigned short* gA1 = wb  + (size_t)(rowBase0 + ar1) * MPAD + as1 * 8;
    const unsigned short* gB  = mTb + (size_t)(colBase0 + br ) * MPAD + bs  * 8;
    unsigned lA0[2], lA1[2], lB[2];
#pragma unroll
    for (int s = 0; s < 2; ++s) {
        lA0[s] = lds_off_of(&Ab[s][ar0][as0 * 8]);
        lA1[s] = lds_off_of(&Ab[s][ar1][as1 * 8]);
        lB[s]  = lds_off_of(&Bb[s][br ][bs  * 8]);
    }

    v8f acc[2][2];
#pragma unroll
    for (int i = 0; i < 2; ++i)
#pragma unroll
        for (int j = 0; j < 2; ++j) acc[i][j] = (v8f)(0.0f);

    async_b128(lA0[0], gA0);
    async_b128(lA1[0], gA1);
    async_b128(lB[0],  gB);

    for (int k = 0; k < MPAD; k += 32) {
        const int cur = (k >> 5) & 1;
        const int nxt = cur ^ 1;
        if (k + 32 < MPAD) {
            async_b128(lA0[nxt], gA0 + k + 32);
            async_b128(lA1[nxt], gA1 + k + 32);
            async_b128(lB[nxt],  gB  + k + 32);
            asm volatile("s_wait_asynccnt 0x3" ::: "memory");
        } else {
            asm volatile("s_wait_asynccnt 0x0" ::: "memory");
        }
        __syncthreads();

        v16bf fa0 = load_frag_a(&Ab[cur][wm * 32 + r][half * 16]);
        v16bf fa1 = load_frag_a(&Ab[cur][wm * 32 + 16 + r][half * 16]);
        v16bf fb0 = load_frag_b(&Bb[cur][wn * 32 + r][half * 16]);
        v16bf fb1 = load_frag_b(&Bb[cur][wn * 32 + 16 + r][half * 16]);
        acc[0][0] = wmma_bf16(fa0, fb0, acc[0][0]);
        acc[0][1] = wmma_bf16(fa0, fb1, acc[0][1]);
        acc[1][0] = wmma_bf16(fa1, fb0, acc[1][0]);
        acc[1][1] = wmma_bf16(fa1, fb1, acc[1][1]);

        __syncthreads();
    }

#pragma unroll
    for (int j = 0; j < 2; ++j) {
        const int gj = colBase0 + wn * 32 + j * 16 + r;
#pragma unroll
        for (int i = 0; i < 2; ++i) {
#pragma unroll
            for (int v = 0; v < 8; ++v) {
                const int gi = rowBase0 + wm * 32 + i * 16 + v + half * 8;
                zhat[(size_t)gi * DDIM + gj] = acc[i][j][v];
            }
        }
    }
}

// ---------------------------------------------------------------------------
extern "C" void kernel_launch(void* const* d_in, const int* in_sizes, int n_in,
                              void* d_out, int out_size, void* d_ws, size_t ws_size,
                              hipStream_t stream) {
    const float* x   = (const float*)d_in[0];   // (8192,256,4,4)
    const float* mem = (const float*)d_in[1];   // (600,4096)

    float* zhat = (float*)d_out;                                 // 8192*4096
    float* wout = (float*)d_out + (size_t)BATCH * DDIM;          // 8192*600

    char* ws = (char*)d_ws;
    unsigned short* zb  = (unsigned short*)ws;  ws += (size_t)BATCH * DDIM * 2; // 64 MiB
    unsigned short* mb  = (unsigned short*)ws;  ws += (size_t)MPAD  * DDIM * 2; //  5 MiB
    unsigned short* mTb = (unsigned short*)ws;  ws += (size_t)DDIM  * MPAD * 2; //  5 MiB
    unsigned short* wbf = (unsigned short*)ws;  ws += (size_t)BATCH * MPAD * 2; // 10 MiB
    float* znorm = (float*)ws;                  ws += (size_t)BATCH * 4;
    float* mnorm = (float*)ws;                  ws += (size_t)MPAD  * 4;

    prep_z_kernel<<<BATCH, 256, 0, stream>>>(x, zb, znorm);
    prep_mem_kernel<<<MPAD, 256, 0, stream>>>(mem, mb, mTb, mnorm);

    dim3 g1(BATCH / 128, MPAD / 64);            // 64 x 10
    gemm1_kernel<<<g1, 256, 0, stream>>>(zb, mb, znorm, mnorm, wout);

    softmax_shrink_kernel<<<BATCH, 256, 0, stream>>>(wout, wbf);

    dim3 g2(BATCH / 128, DDIM / 64);            // 64 x 64
    gemm2_kernel<<<g2, 256, 0, stream>>>(wbf, mTb, zhat);
}
